// DeltaNetGen2DBlock_23459111371404
// MI455X (gfx1250) — compile-verified, matrix operand order checked by reading, not run on hardware
//
#include <hip/hip_runtime.h>
#include <hip/hip_bf16.h>

// ---------------------------------------------------------------------------
// DeltaNet DiT-style block for MI455X (gfx1250, wave32, WMMA).
// Big GEMMs: 128x128 block tiles, double-buffered async global->LDS staging,
// v_wmma_f32_16x16x32_f16 with fp32 accumulate (8 WMMA sub-tiles per wave).
// Delta-rule recurrence: one block per (batch, head), 64x64 state in regs.
// ---------------------------------------------------------------------------

typedef _Float16 half_t;
typedef __attribute__((ext_vector_type(16))) _Float16 v16h;
typedef __attribute__((ext_vector_type(8)))  _Float16 v8h;
typedef __attribute__((ext_vector_type(8)))  float    v8f;
typedef __attribute__((ext_vector_type(4)))  int      v4i_;

#define Bv    2
#define Tv    2048
#define Dv    1024
#define Hv    16
#define DKv   64
#define DVv   64
#define DFFv  4096
#define CONVv 4
#define EPSv  1e-6f
#define ROWS  (Bv * Tv)   // 4096

// ---- CDNA5 async global->LDS path (ASYNCcnt), with portable fallback ------
#if __has_builtin(__builtin_amdgcn_global_load_async_to_lds_b128)
#define HAVE_ASYNC_LDS 1
#endif

static __device__ __forceinline__ void cp16_g2l(const half_t* g, half_t* l) {
#ifdef HAVE_ASYNC_LDS
  // Builtin signature (from hipcc diagnostic): param 1 is int4 in the global
  // (__device__ / addrspace(1)) AS; param 2 is the LDS destination. C-style
  // casts perform the required addrspacecasts (generic->AS1 / generic->AS3).
  __builtin_amdgcn_global_load_async_to_lds_b128(
      (__attribute__((address_space(1))) v4i_*)g,
      (__attribute__((address_space(3))) v4i_*)l, /*offset=*/0, /*cpol=*/0);
#else
  *(v8h*)l = *(const v8h*)g;
#endif
}

static __device__ __forceinline__ void wait_async_lds() {
#ifdef HAVE_ASYNC_LDS
#if __has_builtin(__builtin_amdgcn_s_wait_asynccnt)
  __builtin_amdgcn_s_wait_asynccnt(0);
#else
  asm volatile("s_wait_asynccnt 0" ::: "memory");
#endif
#endif
}

static __device__ __forceinline__ float sigmoidf_(float x) {
  return 1.f / (1.f + __expf(-x));
}

// ---------------------------------------------------------------------------
// mods = silu(c) @ W_ada + b_ada        c:[B,D]  W_ada:[D,6D]  -> mods:[B,6D]
// ---------------------------------------------------------------------------
__global__ void k_mods(const float* __restrict__ c, const float* __restrict__ Wada,
                       const float* __restrict__ bada, float* __restrict__ mods) {
  int j = blockIdx.x * blockDim.x + threadIdx.x;   // coalesced over Wada
  int b = blockIdx.y;
  if (j >= 6 * Dv) return;
  float acc = bada[j];
  const float* cr = c + b * Dv;
  for (int d = 0; d < Dv; ++d) {
    float cv = cr[d];
    acc += (cv * sigmoidf_(cv)) * Wada[(long)d * (6 * Dv) + j];
  }
  mods[b * 6 * Dv + j] = acc;
}

// ---------------------------------------------------------------------------
// Wt[n,k] = (f16) W[k,n]  (transpose + downconvert: both WMMA fragments then
// load two contiguous 16B chunks per lane)
// ---------------------------------------------------------------------------
__global__ void k_transpose_h(const float* __restrict__ W, half_t* __restrict__ Wt,
                              int K, int N) {
  int k = blockIdx.x * 32 + (threadIdx.x & 31);
  int n = blockIdx.y * 8 + (threadIdx.x >> 5);
  if (k < K && n < N) Wt[(long)n * K + k] = (half_t)W[(long)k * N + n];
}

// ---------------------------------------------------------------------------
// LayerNorm (no affine) + modulate -> f16 row.  One block per (b,t) row.
// ---------------------------------------------------------------------------
__global__ void k_ln_mod(const float* __restrict__ x, const float* __restrict__ mods,
                         int shift_off, int scale_off, half_t* __restrict__ out) {
  int row = blockIdx.x;
  int b = row / Tv;
  const float* xr = x + (long)row * Dv;
  __shared__ float red[256];
  __shared__ float s_mu, s_rstd;
  float lv[4];
  float lsum = 0.f;
#pragma unroll
  for (int i = 0; i < 4; ++i) { lv[i] = xr[threadIdx.x + 256 * i]; lsum += lv[i]; }
  red[threadIdx.x] = lsum; __syncthreads();
  for (int s = 128; s > 0; s >>= 1) {
    if ((int)threadIdx.x < s) red[threadIdx.x] += red[threadIdx.x + s];
    __syncthreads();
  }
  if (threadIdx.x == 0) s_mu = red[0] * (1.f / Dv);
  __syncthreads();
  float mu = s_mu, lss = 0.f;
#pragma unroll
  for (int i = 0; i < 4; ++i) { float d = lv[i] - mu; lss += d * d; }
  red[threadIdx.x] = lss; __syncthreads();
  for (int s = 128; s > 0; s >>= 1) {
    if ((int)threadIdx.x < s) red[threadIdx.x] += red[threadIdx.x + s];
    __syncthreads();
  }
  if (threadIdx.x == 0) s_rstd = rsqrtf(red[0] * (1.f / Dv) + EPSv);
  __syncthreads();
  float rstd = s_rstd;
  const float* sh = mods + b * 6 * Dv + shift_off;
  const float* sc = mods + b * 6 * Dv + scale_off;
#pragma unroll
  for (int i = 0; i < 4; ++i) {
    int d = threadIdx.x + 256 * i;
    float xn = (lv[i] - mu) * rstd;
    out[(long)row * Dv + d] = (half_t)(xn * (1.f + sc[d]) + sh[d]);
  }
}

// ---------------------------------------------------------------------------
// Simple WMMA GEMM (one 16x16 tile per wave) — used only for the N=16 beta
// projection where 128-wide tiling doesn't apply.
// ---------------------------------------------------------------------------
__global__ __launch_bounds__(256) void
k_gemm_wmma(const half_t* __restrict__ A, const half_t* __restrict__ Bt,
            float* __restrict__ C, int M, int N, int K) {
  int wave = threadIdx.x >> 5;
  int lane = threadIdx.x & 31;
  int tiles_n = N >> 4;
  int total = (M >> 4) * tiles_n;
  int tile = blockIdx.x * (blockDim.x >> 5) + wave;   // uniform per wave
  if (tile >= total) return;
  int mt = tile / tiles_n;
  int nt = tile % tiles_n;
  int hl = lane & 15, sel = lane >> 4;

  const half_t* arow = A  + (long)(mt * 16 + hl) * K + sel * 8;
  const half_t* brow = Bt + (long)(nt * 16 + hl) * K + sel * 8;

  union Frag { v16h v; v8h h[2]; };
  v8f acc = {};
  for (int kb = 0; kb < K; kb += 32) {
    Frag fa, fb;
    fa.h[0] = *(const v8h*)(arow + kb);
    fa.h[1] = *(const v8h*)(arow + kb + 16);
    fb.h[0] = *(const v8h*)(brow + kb);
    fb.h[1] = *(const v8h*)(brow + kb + 16);
    acc = __builtin_amdgcn_wmma_f32_16x16x32_f16(
        false, fa.v, false, fb.v, (short)0, acc, false, false);
  }
  float* crow = C + (long)(mt * 16 + sel * 8) * N + nt * 16 + hl;
#pragma unroll
  for (int r = 0; r < 8; ++r) crow[(long)r * N] = acc[r];
}

// ---------------------------------------------------------------------------
// Tiled WMMA GEMM: C[M,N](f32) = A[M,K](f16 row-major) * Bt[N,K](f16)^T
//   - 128x128 block tile, BK=32, double-buffered LDS staged via async
//     global->LDS (ASYNCcnt) when available.
//   - 8 waves in 4(M) x 2(N) grid; each wave: 2x4 = 8 16x16 WMMA sub-tiles.
//   - Fragment layout per CDNA5 ISA 7.12.2 (16-bit 16x32):
//       lanes 0-15 : elems 0..7 = K 0..7,  8..15 = K 16..23 (M/N = lane)
//       lanes 16-31: elems 0..7 = K 8..15, 8..15 = K 24..31 (M/N = lane-16)
// ---------------------------------------------------------------------------
#define BM 128
#define BN 128
#define BK 32

__global__ __launch_bounds__(256) void
k_gemm_wmma_tiled(const half_t* __restrict__ A, const half_t* __restrict__ Bt,
                  float* __restrict__ C, int M, int N, int K) {
  __shared__ __align__(16) half_t sA[2][BM][BK];
  __shared__ __align__(16) half_t sB[2][BN][BK];
  int tid = threadIdx.x;
  int wave = tid >> 5, lane = tid & 31;
  int hl = lane & 15, sel = lane >> 4;
  int gn_tiles = N / BN;
  int bm = blockIdx.x / gn_tiles;
  int bn = blockIdx.x % gn_tiles;
  int wm = wave >> 1;   // 0..3 -> 32-row strip
  int wn = wave & 1;    // 0..1 -> 64-col strip

  const half_t* Ab = A  + (long)bm * BM * K;
  const half_t* Bb = Bt + (long)bn * BN * K;

  // Stage one K-slab (128x32 f16 = 8KB each for A and B): 512 16B chunks per
  // tile, 2 per thread. Async writes go straight to LDS (no VGPR round trip).
  auto stage = [&](int buf, int kb) {
#pragma unroll
    for (int i = 0; i < 2; ++i) {
      int cch = tid + 256 * i;
      int row = cch >> 2;
      int ch  = (cch & 3) * 8;           // halves
      cp16_g2l(Ab + (long)row * K + kb + ch, &sA[buf][row][ch]);
      cp16_g2l(Bb + (long)row * K + kb + ch, &sB[buf][row][ch]);
    }
  };

  union Frag { v16h v; v8h h[2]; };
  v8f acc[2][4] = {};

  stage(0, 0);
  wait_async_lds();
  __syncthreads();

  int nk = K / BK;
  for (int s = 0; s < nk; ++s) {
    int buf = s & 1;
    if (s + 1 < nk) stage(buf ^ 1, (s + 1) * BK);

    Frag fa[2], fb[4];
#pragma unroll
    for (int sm = 0; sm < 2; ++sm) {
      int row = wm * 32 + sm * 16 + hl;
      fa[sm].h[0] = *(const v8h*)&sA[buf][row][sel * 8];
      fa[sm].h[1] = *(const v8h*)&sA[buf][row][sel * 8 + 16];
    }
#pragma unroll
    for (int sn = 0; sn < 4; ++sn) {
      int col = wn * 64 + sn * 16 + hl;
      fb[sn].h[0] = *(const v8h*)&sB[buf][col][sel * 8];
      fb[sn].h[1] = *(const v8h*)&sB[buf][col][sel * 8 + 16];
    }
#pragma unroll
    for (int sm = 0; sm < 2; ++sm)
#pragma unroll
      for (int sn = 0; sn < 4; ++sn)
        acc[sm][sn] = __builtin_amdgcn_wmma_f32_16x16x32_f16(
            false, fa[sm].v, false, fb[sn].v, (short)0, acc[sm][sn],
            false, false);

    wait_async_lds();   // next slab fully in LDS before anyone flips buffers
    __syncthreads();
  }

  // C/D layout: VGPR r -> row = tile_m + sel*8 + r, col = tile_n + hl
#pragma unroll
  for (int sm = 0; sm < 2; ++sm) {
#pragma unroll
    for (int sn = 0; sn < 4; ++sn) {
      int row0 = bm * BM + wm * 32 + sm * 16 + sel * 8;
      int col  = bn * BN + wn * 64 + sn * 16 + hl;
      float* cr = C + (long)row0 * N + col;
#pragma unroll
      for (int r = 0; r < 8; ++r) cr[(long)r * N] = acc[sm][sn][r];
    }
  }
}

// ---------------------------------------------------------------------------
// Causal depthwise conv (kernel 4) + SiLU, optional per-head L2 norm.
// ---------------------------------------------------------------------------
__global__ void k_conv(const float* __restrict__ in, const float* __restrict__ w,
                       float* __restrict__ out, int do_norm) {
  int row = blockIdx.x;
  int t = row % Tv;
  __shared__ float sy[Dv];
  __shared__ float sinv[Hv];
#pragma unroll
  for (int i = 0; i < 4; ++i) {
    int d = threadIdx.x + 256 * i;
    float acc = 0.f;
#pragma unroll
    for (int j = 0; j < CONVv; ++j) {
      int tt = t - (CONVv - 1) + j;
      float xv = (tt >= 0) ? in[(long)(row - (CONVv - 1) + j) * Dv + d] : 0.f;
      acc += xv * w[d * CONVv + j];
    }
    sy[d] = acc * sigmoidf_(acc);   // SiLU
  }
  __syncthreads();
  if (do_norm) {
    if ((int)threadIdx.x < Hv) {
      float ss = 0.f;
      for (int cidx = 0; cidx < DKv; ++cidx) {
        float v = sy[threadIdx.x * DKv + cidx];
        ss += v * v;
      }
      sinv[threadIdx.x] = rsqrtf(ss + EPSv);   // l2norm uses sum (not mean)
    }
    __syncthreads();
  }
#pragma unroll
  for (int i = 0; i < 4; ++i) {
    int d = threadIdx.x + 256 * i;
    float y = sy[d];
    if (do_norm) y *= sinv[d >> 6];
    out[(long)row * Dv + d] = y;
  }
}

// ---------------------------------------------------------------------------
// Delta-rule recurrence. One block per (b,h), 64 threads (thread = v column),
// state S[k][v] in 64 registers per thread. Sequential over T.
// ---------------------------------------------------------------------------
__global__ __launch_bounds__(64) void
k_delta(const float* __restrict__ q, const float* __restrict__ k,
        const float* __restrict__ v, const float* __restrict__ betain,
        float* __restrict__ o) {
  int b = blockIdx.x / Hv;
  int h = blockIdx.x % Hv;
  int vi = threadIdx.x;
  float S[DKv];
#pragma unroll
  for (int i = 0; i < DKv; ++i) S[i] = 0.f;
  __shared__ float sq[DKv], sk[DKv];
  const float scale = 0.125f;   // DK^-0.5
  for (int t = 0; t < Tv; ++t) {
    long base = ((long)(b * Tv + t) * Hv + h) * DKv;
    sq[vi] = q[base + vi] * scale;
    sk[vi] = k[base + vi];
    __syncthreads();
    float vt = v[base + vi];
    float beta = sigmoidf_(betain[(long)(b * Tv + t) * Hv + h]);
    float pred = 0.f;
#pragma unroll
    for (int kk = 0; kk < DKv; ++kk) pred += sk[kk] * S[kk];
    float delta = beta * (vt - pred);
    float ov = 0.f;
#pragma unroll
    for (int kk = 0; kk < DKv; ++kk) {
      S[kk] += sk[kk] * delta;
      ov += sq[kk] * S[kk];
    }
    o[base + vi] = ov;
    __syncthreads();
  }
}

// ---------------------------------------------------------------------------
// Per-head RMSNorm of o (mean of squares) * w_onorm -> f16 for the Wo GEMM.
// ---------------------------------------------------------------------------
__global__ void k_onorm(const float* __restrict__ o, const float* __restrict__ w,
                        half_t* __restrict__ oh) {
  int row = blockIdx.x;
  __shared__ float sy[Dv];
  __shared__ float sinv[Hv];
#pragma unroll
  for (int i = 0; i < 4; ++i) {
    int d = threadIdx.x + 256 * i;
    sy[d] = o[(long)row * Dv + d];
  }
  __syncthreads();
  if ((int)threadIdx.x < Hv) {
    float ss = 0.f;
    for (int cidx = 0; cidx < DVv; ++cidx) {
      float v = sy[threadIdx.x * DVv + cidx];
      ss += v * v;
    }
    sinv[threadIdx.x] = rsqrtf(ss * (1.f / DVv) + EPSv);
  }
  __syncthreads();
#pragma unroll
  for (int i = 0; i < 4; ++i) {
    int d = threadIdx.x + 256 * i;
    oh[(long)row * Dv + d] = (half_t)(sy[d] * sinv[d >> 6] * w[d & (DVv - 1)]);
  }
}

// ---------------------------------------------------------------------------
// x1 = x + g_a * attn; then LayerNorm(x1) + modulate(sh_m, sc_m) -> f16 h.
// ---------------------------------------------------------------------------
__global__ void k_resid_ln_mod(const float* __restrict__ x, const float* __restrict__ attn,
                               const float* __restrict__ mods,
                               float* __restrict__ x1, half_t* __restrict__ hh) {
  int row = blockIdx.x;
  int b = row / Tv;
  const float* ga = mods + b * 6 * Dv + 2 * Dv;
  const float* sh = mods + b * 6 * Dv + 3 * Dv;
  const float* sc = mods + b * 6 * Dv + 4 * Dv;
  __shared__ float red[256];
  __shared__ float s_mu, s_rstd;
  float lv[4];
  float lsum = 0.f;
#pragma unroll
  for (int i = 0; i < 4; ++i) {
    int d = threadIdx.x + 256 * i;
    long idx = (long)row * Dv + d;
    lv[i] = x[idx] + ga[d] * attn[idx];
    x1[idx] = lv[i];
    lsum += lv[i];
  }
  red[threadIdx.x] = lsum; __syncthreads();
  for (int s = 128; s > 0; s >>= 1) {
    if ((int)threadIdx.x < s) red[threadIdx.x] += red[threadIdx.x + s];
    __syncthreads();
  }
  if (threadIdx.x == 0) s_mu = red[0] * (1.f / Dv);
  __syncthreads();
  float mu = s_mu, lss = 0.f;
#pragma unroll
  for (int i = 0; i < 4; ++i) { float d = lv[i] - mu; lss += d * d; }
  red[threadIdx.x] = lss; __syncthreads();
  for (int s = 128; s > 0; s >>= 1) {
    if ((int)threadIdx.x < s) red[threadIdx.x] += red[threadIdx.x + s];
    __syncthreads();
  }
  if (threadIdx.x == 0) s_rstd = rsqrtf(red[0] * (1.f / Dv) + EPSv);
  __syncthreads();
  float rstd = s_rstd;
#pragma unroll
  for (int i = 0; i < 4; ++i) {
    int d = threadIdx.x + 256 * i;
    float xn = (lv[i] - mu) * rstd;
    hh[(long)row * Dv + d] = (half_t)(xn * (1.f + sc[d]) + sh[d]);
  }
}

// ---------------------------------------------------------------------------
// g1 = gelu_tanh(m1 + b1) -> f16
// ---------------------------------------------------------------------------
__global__ void k_bias_gelu(const float* __restrict__ m1, const float* __restrict__ b1,
                            half_t* __restrict__ g1, int N) {
  long i = (long)blockIdx.x * 256 + threadIdx.x;
  float xv = m1[i] + b1[i % N];
  float t = tanhf(0.7978845608f * (xv + 0.044715f * xv * xv * xv));
  g1[i] = (half_t)(0.5f * xv * (1.f + t));
}

// ---------------------------------------------------------------------------
// out = x1 + g_m * (m2 + b2)
// ---------------------------------------------------------------------------
__global__ void k_final(const float* __restrict__ x1, const float* __restrict__ m2,
                        const float* __restrict__ b2, const float* __restrict__ mods,
                        float* __restrict__ out) {
  long i = (long)blockIdx.x * 256 + threadIdx.x;
  int d = (int)(i % Dv);
  int row = (int)(i / Dv);
  int b = row / Tv;
  float gm = mods[b * 6 * Dv + 5 * Dv + d];
  out[i] = x1[i] + gm * (m2[i] + b2[d]);
}

// ---------------------------------------------------------------------------
extern "C" void kernel_launch(void* const* d_in, const int* in_sizes, int n_in,
                              void* d_out, int out_size, void* d_ws, size_t ws_size,
                              hipStream_t stream) {
  (void)in_sizes; (void)n_in; (void)out_size; (void)ws_size;
  const float* x    = (const float*)d_in[0];
  const float* c    = (const float*)d_in[1];
  const float* Wada = (const float*)d_in[2];
  const float* bada = (const float*)d_in[3];
  const float* Wq   = (const float*)d_in[4];
  const float* Wk   = (const float*)d_in[5];
  const float* Wv   = (const float*)d_in[6];
  const float* Wb   = (const float*)d_in[7];
  const float* wcq  = (const float*)d_in[8];
  const float* wck  = (const float*)d_in[9];
  const float* wcv  = (const float*)d_in[10];
  const float* won  = (const float*)d_in[11];
  const float* Wo   = (const float*)d_in[12];
  const float* W1   = (const float*)d_in[13];
  const float* b1   = (const float*)d_in[14];
  const float* W2   = (const float*)d_in[15];
  const float* b2   = (const float*)d_in[16];
  float* out = (float*)d_out;
  char* ws = (char*)d_ws;

  // ---- scratch layout (phase-aliased arena; peak ~155 MB) ----
  size_t off = 0;
  auto alloc = [&](size_t bytes) {
    size_t o = off; off = (off + bytes + 255) & ~(size_t)255; return o;
  };
  const size_t MB = 1024 * 1024;
  size_t o_mods = alloc((size_t)Bv * 6 * Dv * 4);
  size_t o_wqt  = alloc((size_t)Dv * Dv * 2);
  size_t o_wkt  = alloc((size_t)Dv * Dv * 2);
  size_t o_wvt  = alloc((size_t)Dv * Dv * 2);
  size_t o_wbt  = alloc((size_t)Hv * Dv * 2);
  size_t o_wot  = alloc((size_t)Dv * Dv * 2);
  size_t o_w1t  = alloc((size_t)DFFv * Dv * 2);
  size_t o_w2t  = alloc((size_t)Dv * DFFv * 2);
  size_t o_xm   = alloc((size_t)ROWS * Dv * 2);
  size_t o_x1   = alloc((size_t)ROWS * Dv * 4);
  size_t o_hh   = alloc((size_t)ROWS * Dv * 2);
  size_t o_ar   = alloc(98 * MB);
  // arena phases:
  size_t o_qin  = o_ar + 0;        // [ROWS,D] f32  (dead after conv)
  size_t o_kin  = o_ar + 16 * MB;
  size_t o_vin  = o_ar + 32 * MB;
  size_t o_bin  = o_ar + 48 * MB;  // [ROWS,H] f32  (live through delta)
  size_t o_q    = o_ar + 49 * MB;  // (dead after delta)
  size_t o_k    = o_ar + 65 * MB;
  size_t o_v    = o_ar + 81 * MB;
  size_t o_o    = o_ar + 0;        // over qin
  size_t o_oh   = o_ar + 16 * MB;  // over kin
  size_t o_attn = o_ar + 32 * MB;  // over vin
  size_t o_m1   = o_ar + 0;        // 64MB, over qin..q (dead)
  size_t o_g1   = o_ar + 64 * MB;  // 32MB, over q/k/v (dead)
  size_t o_m2   = o_ar + 0;        // over m1 (dead after gelu)

  // ---- 1. adaLN modulation vector ----
  dim3 gm(6 * Dv / 256, Bv);
  k_mods<<<gm, 256, 0, stream>>>(c, Wada, bada, (float*)(ws + o_mods));

  // ---- 2. weight transpose + f16 downconvert ----
  auto launch_t = [&](const float* W, size_t o, int K, int N) {
    dim3 g(K / 32, N / 8);
    k_transpose_h<<<g, 256, 0, stream>>>(W, (half_t*)(ws + o), K, N);
  };
  launch_t(Wq, o_wqt, Dv, Dv);
  launch_t(Wk, o_wkt, Dv, Dv);
  launch_t(Wv, o_wvt, Dv, Dv);
  launch_t(Wb, o_wbt, Dv, Hv);
  launch_t(Wo, o_wot, Dv, Dv);
  launch_t(W1, o_w1t, Dv, DFFv);
  launch_t(W2, o_w2t, DFFv, Dv);

  // ---- 3. LN + modulate (attn branch) -> xm f16 ----
  k_ln_mod<<<ROWS, 256, 0, stream>>>(x, (const float*)(ws + o_mods), 0, Dv,
                                     (half_t*)(ws + o_xm));

  // ---- WMMA GEMM helper (tiled when dims allow, simple otherwise) ----
  auto launch_gemm = [&](size_t oa, size_t ob, size_t oc, int M, int N, int K) {
    if ((M % BM == 0) && (N % BN == 0) && (K % BK == 0)) {
      int grid = (M / BM) * (N / BN);
      k_gemm_wmma_tiled<<<grid, 256, 0, stream>>>((const half_t*)(ws + oa),
                                                  (const half_t*)(ws + ob),
                                                  (float*)(ws + oc), M, N, K);
    } else {
      int total = (M / 16) * (N / 16);
      int grid = (total + 7) / 8;
      k_gemm_wmma<<<grid, 256, 0, stream>>>((const half_t*)(ws + oa),
                                            (const half_t*)(ws + ob),
                                            (float*)(ws + oc), M, N, K);
    }
  };

  // ---- 4. Q/K/V/beta projections ----
  launch_gemm(o_xm, o_wqt, o_qin, ROWS, Dv, Dv);
  launch_gemm(o_xm, o_wkt, o_kin, ROWS, Dv, Dv);
  launch_gemm(o_xm, o_wvt, o_vin, ROWS, Dv, Dv);
  launch_gemm(o_xm, o_wbt, o_bin, ROWS, Hv, Dv);   // N=16 -> simple kernel

  // ---- 5. short conv + SiLU (+ per-head l2norm for q,k) ----
  k_conv<<<ROWS, 256, 0, stream>>>((const float*)(ws + o_qin), wcq, (float*)(ws + o_q), 1);
  k_conv<<<ROWS, 256, 0, stream>>>((const float*)(ws + o_kin), wck, (float*)(ws + o_k), 1);
  k_conv<<<ROWS, 256, 0, stream>>>((const float*)(ws + o_vin), wcv, (float*)(ws + o_v), 0);

  // ---- 6. delta-rule recurrence ----
  k_delta<<<Bv * Hv, 64, 0, stream>>>((const float*)(ws + o_q), (const float*)(ws + o_k),
                                      (const float*)(ws + o_v), (const float*)(ws + o_bin),
                                      (float*)(ws + o_o));

  // ---- 7. per-head output RMSNorm -> f16 ----
  k_onorm<<<ROWS, 256, 0, stream>>>((const float*)(ws + o_o), won, (half_t*)(ws + o_oh));

  // ---- 8. output projection ----
  launch_gemm(o_oh, o_wot, o_attn, ROWS, Dv, Dv);

  // ---- 9. residual + LN + modulate (mlp branch) ----
  k_resid_ln_mod<<<ROWS, 256, 0, stream>>>(x, (const float*)(ws + o_attn),
                                           (const float*)(ws + o_mods),
                                           (float*)(ws + o_x1), (half_t*)(ws + o_hh));

  // ---- 10. MLP ----
  launch_gemm(o_hh, o_w1t, o_m1, ROWS, DFFv, Dv);
  long n1 = (long)ROWS * DFFv;
  k_bias_gelu<<<(int)(n1 / 256), 256, 0, stream>>>((const float*)(ws + o_m1), b1,
                                                   (half_t*)(ws + o_g1), DFFv);
  launch_gemm(o_g1, o_w2t, o_m2, ROWS, Dv, DFFv);

  // ---- 11. final residual ----
  long n2 = (long)ROWS * Dv;
  k_final<<<(int)(n2 / 256), 256, 0, stream>>>((const float*)(ws + o_x1),
                                               (const float*)(ws + o_m2), b2,
                                               (const float*)(ws + o_mods), out);
}
